// MonarchMixerSequenceMixing_25391846654019
// MI455X (gfx1250) — matrix operand order repaired and checked by reading, not compiled, not run on hardware
//
#include <hip/hip_runtime.h>

typedef _Float16 h8   __attribute__((ext_vector_type(8)));
typedef _Float16 v16h __attribute__((ext_vector_type(16)));
typedef float    v8f  __attribute__((ext_vector_type(8)));
typedef unsigned int v4u __attribute__((ext_vector_type(4)));
typedef int      v4i  __attribute__((ext_vector_type(4)));
typedef int      v8i  __attribute__((ext_vector_type(8)));

#define BB   8
#define LL   4096
#define DD   768
#define ND   (3 * DD)          // 2304
#define KPAD 64                // zero front-pad for Toeplitz gathers
#define LP   (LL + KPAD)       // padded filter plane length (4160)
#define NDELTA 256             // distinct Toeplitz diagonal offsets (delta/16)

// ---------------------------------------------------------------------------
// f32 -> f16 conversion
// ---------------------------------------------------------------------------
__global__ void cvt_f32_f16(const float* __restrict__ in, _Float16* __restrict__ out, int n) {
  int i = blockIdx.x * blockDim.x + threadIdx.x;
  if (i < n) out[i] = (_Float16)in[i];
}

// ---------------------------------------------------------------------------
// WMMA helper
// ---------------------------------------------------------------------------
__device__ __forceinline__ v8f wmma_f16(v16h a, v16h b, v8f c) {
  return __builtin_amdgcn_wmma_f32_16x16x32_f16(false, a, false, b, (short)0, c, false, false);
}

// ---------------------------------------------------------------------------
// WMMA GEMM: C[M,N] = A[M,K] * W[N,K]^T + bias[N]
// One wave per 16x16 output tile. A,W f16 row-major; bias f32.
// ---------------------------------------------------------------------------
__global__ __launch_bounds__(32)
void gemm_out_h16(const _Float16* __restrict__ A, const _Float16* __restrict__ W,
                  const float* __restrict__ bias, _Float16* __restrict__ C,
                  int M, int N, int K) {
  int ntiles_n = N >> 4;
  int tile = blockIdx.x;
  int m0 = (tile / ntiles_n) << 4;
  int n0 = (tile % ntiles_n) << 4;
  int lane = threadIdx.x;
  int mi = lane & 15, hh = lane >> 4;

  const _Float16* arow = A + (size_t)(m0 + mi) * K + 8 * hh;
  const _Float16* brow = W + (size_t)(n0 + mi) * K + 8 * hh;
  v8f acc = {};
  for (int k0 = 0; k0 < K; k0 += 32) {
    h8 a0 = *(const h8*)(arow + k0);
    h8 a1 = *(const h8*)(arow + k0 + 16);
    h8 b0 = *(const h8*)(brow + k0);
    h8 b1 = *(const h8*)(brow + k0 + 16);
    v16h av = __builtin_shufflevector(a0, a1, 0,1,2,3,4,5,6,7,8,9,10,11,12,13,14,15);
    v16h bv = __builtin_shufflevector(b0, b1, 0,1,2,3,4,5,6,7,8,9,10,11,12,13,14,15);
    acc = wmma_f16(av, bv, acc);
  }
  float bcol = bias[n0 + mi];
  #pragma unroll
  for (int r = 0; r < 8; ++r) {
    int row = m0 + r + 8 * hh;
    C[(size_t)row * N + n0 + mi] = (_Float16)(acc[r] + bcol);
  }
}

__global__ __launch_bounds__(32)
void gemm_out_f32(const _Float16* __restrict__ A, const _Float16* __restrict__ W,
                  const float* __restrict__ bias, float* __restrict__ C,
                  int M, int N, int K) {
  int ntiles_n = N >> 4;
  int tile = blockIdx.x;
  int m0 = (tile / ntiles_n) << 4;
  int n0 = (tile % ntiles_n) << 4;
  int lane = threadIdx.x;
  int mi = lane & 15, hh = lane >> 4;

  const _Float16* arow = A + (size_t)(m0 + mi) * K + 8 * hh;
  const _Float16* brow = W + (size_t)(n0 + mi) * K + 8 * hh;
  v8f acc = {};
  for (int k0 = 0; k0 < K; k0 += 32) {
    h8 a0 = *(const h8*)(arow + k0);
    h8 a1 = *(const h8*)(arow + k0 + 16);
    h8 b0 = *(const h8*)(brow + k0);
    h8 b1 = *(const h8*)(brow + k0 + 16);
    v16h av = __builtin_shufflevector(a0, a1, 0,1,2,3,4,5,6,7,8,9,10,11,12,13,14,15);
    v16h bv = __builtin_shufflevector(b0, b1, 0,1,2,3,4,5,6,7,8,9,10,11,12,13,14,15);
    acc = wmma_f16(av, bv, acc);
  }
  float bcol = bias[n0 + mi];
  #pragma unroll
  for (int r = 0; r < 8; ++r) {
    int row = m0 + r + 8 * hh;
    C[(size_t)row * N + n0 + mi] = acc[r] + bcol;
  }
}

// ---------------------------------------------------------------------------
// Depthwise conv (k=3, causal pad 2) + split + gate -> v, x2 in (B,D,L) f16
// ---------------------------------------------------------------------------
__global__ void shortconv_gate(const _Float16* __restrict__ x,
                               const float* __restrict__ cw, const float* __restrict__ cb,
                               _Float16* __restrict__ v16, _Float16* __restrict__ x2_16) {
  int idx = blockIdx.x * blockDim.x + threadIdx.x;
  if (idx >= BB * LL * DD) return;
  int d = idx % DD;
  int t = (idx / DD) % LL;
  int b = idx / (DD * LL);
  float xc[3];
  #pragma unroll
  for (int g = 0; g < 3; ++g) {
    int c = g * DD + d;
    float acc = cb[c];
    #pragma unroll
    for (int k = 0; k < 3; ++k) {
      int tt = t - 2 + k;
      if (tt >= 0)
        acc += cw[c * 3 + k] * (float)x[((size_t)(b * LL + tt)) * ND + c];
    }
    xc[g] = acc;
  }
  size_t p = ((size_t)(b * DD + d)) * LL + t;
  v16[p]   = (_Float16)(xc[2] * xc[0]);
  x2_16[p] = (_Float16)xc[1];
}

// ---------------------------------------------------------------------------
// Implicit Hyena filter: k[d, t] with decay; stored (D, LP) f16, front KPAD zeros
// ---------------------------------------------------------------------------
__global__ void filtergen(const float* __restrict__ fw0, const float* __restrict__ fb0,
                          const float* __restrict__ freq,
                          const float* __restrict__ fw1, const float* __restrict__ fb1,
                          const float* __restrict__ fw2, const float* __restrict__ fb2,
                          const float* __restrict__ fw3, _Float16* __restrict__ k16) {
  int i = blockIdx.x * blockDim.x + threadIdx.x;
  if (i >= LL) return;
  float ti   = (float)i / (float)(LL - 1);
  float wang = 2.0f * 3.14159265358979323846f * (float)i / (float)LL;
  float fz   = 1e-4f;
  float z0 = ti, z1 = cosf(fz * wang), z2 = -sinf(fz * wang);
  float ha[16], hb[16];
  #pragma unroll
  for (int o = 0; o < 16; ++o)
    ha[o] = sinf(freq[o] * (z0 * fw0[o * 3 + 0] + z1 * fw0[o * 3 + 1] + z2 * fw0[o * 3 + 2] + fb0[o]));
  #pragma unroll
  for (int o = 0; o < 16; ++o) {
    float a = fb1[o];
    #pragma unroll
    for (int j = 0; j < 16; ++j) a += ha[j] * fw1[o * 16 + j];
    hb[o] = sinf(freq[o] * a);
  }
  #pragma unroll
  for (int o = 0; o < 16; ++o) {
    float a = fb2[o];
    #pragma unroll
    for (int j = 0; j < 16; ++j) a += hb[j] * fw2[o * 16 + j];
    ha[o] = sinf(freq[o] * a);
  }
  const float min_dec = -3.0701134573253944f;   // log(0.01)/1.5
  const float max_dec = -15.350567286626972f;   // log(0.01)/0.3
  for (int d = 0; d < DD; ++d) {
    float a = 0.f;
    #pragma unroll
    for (int j = 0; j < 16; ++j) a += ha[j] * fw3[d * 16 + j];
    float delta = min_dec + (max_dec - min_dec) * (float)d / (float)(DD - 1);
    float kv = a * expf(-ti * fabsf(delta));
    k16[(size_t)d * LP + KPAD + i] = (_Float16)kv;
    if (i < KPAD) k16[(size_t)d * LP + i] = (_Float16)0.f;   // zero front pad
  }
}

// ---------------------------------------------------------------------------
// Tensor Data Mover: DMA one padded filter plane (LP f16 elements) into LDS@0.
// D# packed per CDNA5 ISA ch.8: group0 {count=1, lds_addr=0, global_addr, type=2},
// group1 {data_size=2B, tensor_dim0=LP, tile_dim0=LP, 1 row}.
// ---------------------------------------------------------------------------
__device__ __forceinline__ void tdm_load_plane_to_lds(const _Float16* kp) {
  unsigned long long ga = (unsigned long long)(uintptr_t)kp;
  v4u g0;
  g0.x = 1u;                                              // count=1
  g0.y = 0u;                                              // lds_addr = 0
  g0.z = (unsigned)(ga & 0xffffffffu);                    // global_addr[31:0]
  g0.w = (unsigned)((ga >> 32) & 0x01ffffffu) | (2u << 30); // ga[56:32] | type=2
  v8i g1;
  g1[0] = 1 << 16;                    // data_size=1 (2-byte elements)
  g1[1] = (LP & 0xffff) << 16;        // tensor_dim0[15:0]
  g1[2] = (LP >> 16) | (1 << 16);     // tensor_dim0[31:16] | tensor_dim1=1
  g1[3] = (LP & 0xffff) << 16;        // tile_dim0 = LP
  g1[4] = 1;                          // tile_dim1 = 1
  g1[5] = LP;                         // tensor_dim0_stride[31:0]
  g1[6] = 0;
  g1[7] = 0;
  v4i gz = {0, 0, 0, 0};
#if defined(__clang_major__) && (__clang_major__ >= 23)
  v8i gz8 = {0, 0, 0, 0, 0, 0, 0, 0};
  __builtin_amdgcn_tensor_load_to_lds(g0, g1, gz, gz, gz8, 0);
#else
  __builtin_amdgcn_tensor_load_to_lds(g0, g1, gz, gz, 0);
#endif
}

// ---------------------------------------------------------------------------
// Pre-swizzle Toeplitz A-fragments into the WMMA per-lane register layout.
// afrag[(d*256 + delta/16)*32 + lane][0..15] = A-tile elements for that lane.
// One block per channel d; TDM stages the k plane into LDS once.
// ---------------------------------------------------------------------------
__global__ __launch_bounds__(256)
void build_afrag(const _Float16* __restrict__ k16, _Float16* __restrict__ afrag) {
  __shared__ __attribute__((aligned(16))) _Float16 kl[LP];
  int d = blockIdx.x;
  const _Float16* kp = k16 + (size_t)d * LP;
  if (threadIdx.x < 32) {
    tdm_load_plane_to_lds(kp);            // tensor_load_to_lds -> LDS offset 0
    __builtin_amdgcn_s_wait_tensorcnt(0);
  }
  __syncthreads();

  int lane = threadIdx.x & 31;
  int wave = threadIdx.x >> 5;
  int mi = lane & 15, hh = lane >> 4;
  for (int j = wave; j < NDELTA; j += 8) {
    v16h av;
    int abase = KPAD + (j << 4) + mi;
    #pragma unroll
    for (int e = 0; e < 16; ++e) {
      int Ke = 16 * hh + e;
      av[e] = kl[abase - Ke];
    }
    _Float16* dst = afrag + (((size_t)d * NDELTA + j) * 32 + lane) * 16;
    *(h8*)(dst)     = __builtin_shufflevector(av, av, 0,1,2,3,4,5,6,7);
    *(h8*)(dst + 8) = __builtin_shufflevector(av, av, 8,9,10,11,12,13,14,15);
  }
}

// ---------------------------------------------------------------------------
// Long causal conv, fragment path: y[b,d,t] = sum_{s<=t} k[d,t-s] v[b,d,s].
// One wave handles TWO t-tiles (t0a, t0a+16) sharing the B(v) stream.
// Inner loop: 4x aligned b128 A-fragment loads + 2x b128 B loads + 2 WMMA.
// ---------------------------------------------------------------------------
__global__ __launch_bounds__(32)
void hyena_longconv_af(const _Float16* __restrict__ afrag, const _Float16* __restrict__ v16,
                       const _Float16* __restrict__ x2_16, const float* __restrict__ d_bias,
                       _Float16* __restrict__ yT) {
  int d   = blockIdx.x;
  int t0a = blockIdx.y << 5;
  int lane = threadIdx.x;
  int mi = lane & 15, hh = lane >> 4;
  bool active = mi < BB;
  const _Float16* vplane  = v16   + ((size_t)mi * DD + d) * LL;
  const _Float16* x2plane = x2_16 + ((size_t)mi * DD + d) * LL;
  const _Float16* abase   = afrag + ((size_t)d * NDELTA * 32 + lane) * 16;

  v8f accA = {}, accB = {};
  for (int s0 = 0; s0 <= t0a; s0 += 32) {
    int ja = (t0a - s0) >> 4;
    const _Float16* pa = abase + (size_t)ja * (32 * 16);
    const _Float16* pb = pa + (32 * 16);
    h8 a0 = *(const h8*)(pa);
    h8 a1 = *(const h8*)(pa + 8);
    h8 c0 = *(const h8*)(pb);
    h8 c1 = *(const h8*)(pb + 8);
    v16h avA = __builtin_shufflevector(a0, a1, 0,1,2,3,4,5,6,7,8,9,10,11,12,13,14,15);
    v16h avB = __builtin_shufflevector(c0, c1, 0,1,2,3,4,5,6,7,8,9,10,11,12,13,14,15);
    v16h bv = {};
    if (active) {
      h8 b0 = *(const h8*)(vplane + s0 + 16 * hh);
      h8 b1 = *(const h8*)(vplane + s0 + 16 * hh + 8);
      bv = __builtin_shufflevector(b0, b1, 0,1,2,3,4,5,6,7,8,9,10,11,12,13,14,15);
      __builtin_prefetch((const void*)(vplane + s0 + 256), 0, 0);  // global_prefetch_b8
    }
    accA = wmma_f16(avA, bv, accA);
    accB = wmma_f16(avB, bv, accB);
  }

  if (active) {
    float db = d_bias[d];
    #pragma unroll
    for (int half = 0; half < 2; ++half) {
      int t0 = t0a + 16 * half;
      v8f acc = half ? accB : accA;
      h8 vv  = *(const h8*)(vplane  + t0 + 8 * hh);
      h8 xx2 = *(const h8*)(x2plane + t0 + 8 * hh);
      #pragma unroll
      for (int r = 0; r < 8; ++r) {
        int t = t0 + 8 * hh + r;
        float y = (acc[r] + (float)vv[r] * db) * (float)xx2[r];
        yT[((size_t)mi * LL + t) * DD + d] = (_Float16)y;
      }
    }
  }
}

// ---------------------------------------------------------------------------
// Fallback long conv (LDS Toeplitz gather), used only if ws too small for afrag
// ---------------------------------------------------------------------------
__global__ __launch_bounds__(32)
void hyena_longconv(const _Float16* __restrict__ k16, const _Float16* __restrict__ v16,
                    const _Float16* __restrict__ x2_16, const float* __restrict__ d_bias,
                    _Float16* __restrict__ yT) {
  __shared__ __attribute__((aligned(16))) _Float16 kl[LP];
  int d  = blockIdx.x;
  int t0 = blockIdx.y << 4;
  int lane = threadIdx.x;
  const _Float16* kp = k16 + (size_t)d * LP;
  for (int i = lane * 8; i < LP; i += 32 * 8)
    *(h8*)(kl + i) = *(const h8*)(kp + i);
  __syncthreads();

  int mi = lane & 15, hh = lane >> 4;
  bool active = mi < BB;
  const _Float16* vplane  = v16   + ((size_t)mi * DD + d) * LL;
  const _Float16* x2plane = x2_16 + ((size_t)mi * DD + d) * LL;

  v8f acc = {};
  int send = t0 + 15;
  for (int s0 = 0; s0 <= send; s0 += 32) {
    v16h av;
    int abase = KPAD + t0 + mi - s0;
    #pragma unroll
    for (int e = 0; e < 16; ++e) {
      int Ke = 16 * hh + e;
      av[e] = kl[abase - Ke];
    }
    v16h bv = {};
    if (active) {
      h8 b0 = *(const h8*)(vplane + s0 + 16 * hh);
      h8 b1 = *(const h8*)(vplane + s0 + 16 * hh + 8);
      bv = __builtin_shufflevector(b0, b1, 0,1,2,3,4,5,6,7,8,9,10,11,12,13,14,15);
    }
    acc = wmma_f16(av, bv, acc);
  }

  if (active) {
    float db = d_bias[d];
    h8 vv  = *(const h8*)(vplane  + t0 + 8 * hh);
    h8 xx2 = *(const h8*)(x2plane + t0 + 8 * hh);
    #pragma unroll
    for (int r = 0; r < 8; ++r) {
      int t = t0 + 8 * hh + r;
      float y = (acc[r] + (float)vv[r] * db) * (float)xx2[r];
      yT[((size_t)mi * LL + t) * DD + d] = (_Float16)y;
    }
  }
}

// ---------------------------------------------------------------------------
// Host launch
// ---------------------------------------------------------------------------
extern "C" void kernel_launch(void* const* d_in, const int* in_sizes, int n_in,
                              void* d_out, int out_size, void* d_ws, size_t ws_size,
                              hipStream_t stream) {
  const float* u      = (const float*)d_in[0];
  const float* in_w   = (const float*)d_in[1];
  const float* in_b   = (const float*)d_in[2];
  const float* conv_w = (const float*)d_in[3];
  const float* conv_b = (const float*)d_in[4];
  const float* f_w0   = (const float*)d_in[5];
  const float* f_b0   = (const float*)d_in[6];
  const float* freq   = (const float*)d_in[7];
  const float* f_w1   = (const float*)d_in[8];
  const float* f_b1   = (const float*)d_in[9];
  const float* f_w2   = (const float*)d_in[10];
  const float* f_b2   = (const float*)d_in[11];
  const float* f_w3   = (const float*)d_in[12];
  const float* d_bias = (const float*)d_in[13];
  const float* out_w  = (const float*)d_in[14];
  const float* out_b  = (const float*)d_in[15];
  float* out = (float*)d_out;

  char* ws = (char*)d_ws;
  size_t off = 0;
  auto alloc = [&](size_t bytes) -> void* {
    void* p = ws + off;
    off = (off + bytes + 255) & ~(size_t)255;
    return p;
  };
  const size_t M = (size_t)BB * LL;                       // 32768
  _Float16* u16     = (_Float16*)alloc(M * DD * 2);
  _Float16* in_w16  = (_Float16*)alloc((size_t)ND * DD * 2);
  _Float16* out_w16 = (_Float16*)alloc((size_t)DD * DD * 2);
  _Float16* x16     = (_Float16*)alloc(M * ND * 2);
  _Float16* v16     = (_Float16*)alloc(M * DD * 2 + 256); // tail pad for over-reads
  _Float16* x2_16   = (_Float16*)alloc(M * DD * 2);
  _Float16* k16     = (_Float16*)alloc((size_t)DD * LP * 2);
  _Float16* yT16    = (_Float16*)alloc(M * DD * 2);
  size_t base_need = off;
  _Float16* afrag   = (_Float16*)alloc((size_t)DD * NDELTA * 32 * 16 * 2); // 201 MB
  bool useAf = (off <= ws_size);
  if (base_need > ws_size) return;   // workspace too small; bail deterministically

  // 1) f32 -> f16 conversions
  {
    int n = (int)(M * DD);
    cvt_f32_f16<<<(n + 255) / 256, 256, 0, stream>>>(u, u16, n);
    n = ND * DD;
    cvt_f32_f16<<<(n + 255) / 256, 256, 0, stream>>>(in_w, in_w16, n);
    n = DD * DD;
    cvt_f32_f16<<<(n + 255) / 256, 256, 0, stream>>>(out_w, out_w16, n);
  }

  // 2) in_proj: x = u @ in_w^T + in_b   (f16 out)
  gemm_out_h16<<<dim3((unsigned)((M / 16) * (ND / 16))), 32, 0, stream>>>(
      u16, in_w16, in_b, x16, (int)M, ND, DD);

  // 3) depthwise conv + gate
  {
    int n = BB * LL * DD;
    shortconv_gate<<<(n + 255) / 256, 256, 0, stream>>>(x16, conv_w, conv_b, v16, x2_16);
  }

  // 4) implicit filter
  filtergen<<<LL / 256, 256, 0, stream>>>(f_w0, f_b0, freq, f_w1, f_b1, f_w2, f_b2, f_w3, k16);

  // 5) long causal conv (Toeplitz WMMA) + bias/gate epilogue, transposed store
  if (useAf) {
    build_afrag<<<DD, 256, 0, stream>>>(k16, afrag);
    hyena_longconv_af<<<dim3(DD, LL / 32), 32, 0, stream>>>(afrag, v16, x2_16, d_bias, yT16);
  } else {
    hyena_longconv<<<dim3(DD, LL / 16), 32, 0, stream>>>(k16, v16, x2_16, d_bias, yT16);
  }

  // 6) out_proj: out = yT @ out_w^T + out_b  (f32 out)
  gemm_out_f32<<<dim3((unsigned)((M / 16) * (DD / 16))), 32, 0, stream>>>(
      yT16, out_w16, out_b, out, (int)M, DD, DD);
}
